// factor_learning_52364241273286
// MI455X (gfx1250) — compile-verified, matrix operand order checked by reading, not run on hardware
//
#include <hip/hip_runtime.h>
#include <math.h>

typedef __attribute__((ext_vector_type(16))) _Float16 v16h;
typedef __attribute__((ext_vector_type(8)))  float    v8f;

#define DIMC     512
#define HWC      8
#define BATCH    64
#define LSEQ     128
#define D_STATE  16
#define D_CONVK  4
#define D_INNER  1024
#define DT_RANK  32
#define MROWS    (BATCH*LSEQ)   // 8192

// ---------------------------------------------------------------------------
// Build fts0 (B, L, DIM) row-major: reference reshapes (B,512,8,8)->(B,64,512)
// so per batch it is literally spt flat (32768 floats) then qry flat.
// ---------------------------------------------------------------------------
__global__ void concat_kernel(const float* __restrict__ spt,
                              const float* __restrict__ qry,
                              float* __restrict__ fts0) {
  int id = blockIdx.x * blockDim.x + threadIdx.x;   // exactly B*65536 threads
  int b = id >> 16;
  int r = id & 65535;
  float v = (r < 32768) ? spt[(size_t)b * 32768 + r]
                        : qry[(size_t)b * 32768 + (r - 32768)];
  fts0[id] = v;
}

// ---------------------------------------------------------------------------
// Pack an (N x K) f32 weight (element at n*ldn + k*strk) into f16 tiles in the
// CDNA5 WMMA 16-bit B-operand layout: per (ntile,kstep) a 32-lane block where
// lane holds column n = (lane&15), kgroup = lane>>4, and the 16 halves are
// k = kstep*32 + kg*8 + {0..7} and + {16..23}.
// ---------------------------------------------------------------------------
__global__ void pack_w_kernel(const float* __restrict__ W, _Float16* __restrict__ Wp,
                              int N, int K, int ldn, int strk) {
  int idx = blockIdx.x * blockDim.x + threadIdx.x;
  int total = (N >> 4) * (K >> 5) * 32;
  if (idx >= total) return;
  int lane = idx & 31;
  int t    = idx >> 5;
  int ksteps = K >> 5;
  int ntile = t / ksteps;
  int kstep = t - ntile * ksteps;
  int n  = ntile * 16 + (lane & 15);
  int kg = lane >> 4;
  const float* src = W + (size_t)n * ldn + (size_t)(kstep * 32 + kg * 8) * strk;
  _Float16* o = Wp + (size_t)idx * 16;
#pragma unroll
  for (int i = 0; i < 8; ++i) o[i]     = (_Float16)src[(size_t)i * strk];
#pragma unroll
  for (int i = 0; i < 8; ++i) o[8 + i] = (_Float16)src[(size_t)(16 + i) * strk];
}

// ---------------------------------------------------------------------------
// WMMA GEMM with 4-wide N register blocking: each wave computes a 16x64 strip
// (4 accumulators). The A fragment is loaded/converted ONCE per 32-wide k-step
// and reused by 4 v_wmma issues -> 4x less A traffic, amortized f32->f16 cvt.
// `shift/use_seq` remap A rows for the 3-tap SAME conv (edge rows -> zeros).
// act: 0=none, 1=softplus.
// ---------------------------------------------------------------------------
__global__ void wmma_gemm_kernel(const float* __restrict__ A,
                                 const _Float16* __restrict__ Wp,
                                 float* __restrict__ C,
                                 const float* __restrict__ bias,
                                 int M, int N, int K, int lda,
                                 int shift, int use_seq, int accum, int act) {
  const int lane = threadIdx.x & 31;
  const int wave = threadIdx.x >> 5;
  const int nquads = N >> 6;                 // 16x64 strips along N
  int tile  = blockIdx.x * 4 + wave;
  int mtile = tile / nquads;
  int nq    = tile - mtile * nquads;
  int ntile0 = nq << 2;

  const int ml = lane & 15;
  const int kg = lane >> 4;
  int row = mtile * 16 + ml;
  int srow = row;
  bool valid = true;
  if (use_seq) {
    int l = (row & (LSEQ - 1)) + shift;
    valid = (l >= 0) && (l < LSEQ);
    srow = row + shift;
  }

  const int ksteps = K >> 5;
  const v16h* bp = (const v16h*)Wp;
  const size_t nstride = (size_t)ksteps * 32;   // v16h stride between n-tiles

  v8f acc[4] = {};
  for (int s = 0; s < ksteps; ++s) {
    v16h af;
#pragma unroll
    for (int i = 0; i < 16; ++i) af[i] = (_Float16)0.0f;
    if (valid) {
      const float* ap = A + (size_t)srow * lda + s * 32 + kg * 8;
      float4 a0 = *(const float4*)(ap + 0);
      float4 a1 = *(const float4*)(ap + 4);
      float4 a2 = *(const float4*)(ap + 16);
      float4 a3 = *(const float4*)(ap + 20);
      af[0]  = (_Float16)a0.x; af[1]  = (_Float16)a0.y;
      af[2]  = (_Float16)a0.z; af[3]  = (_Float16)a0.w;
      af[4]  = (_Float16)a1.x; af[5]  = (_Float16)a1.y;
      af[6]  = (_Float16)a1.z; af[7]  = (_Float16)a1.w;
      af[8]  = (_Float16)a2.x; af[9]  = (_Float16)a2.y;
      af[10] = (_Float16)a2.z; af[11] = (_Float16)a2.w;
      af[12] = (_Float16)a3.x; af[13] = (_Float16)a3.y;
      af[14] = (_Float16)a3.z; af[15] = (_Float16)a3.w;
    }
    const v16h* bbase = bp + ((size_t)ntile0 * nstride + (size_t)s * 32 + lane);
    v16h bf0 = bbase[0];
    v16h bf1 = bbase[nstride];
    v16h bf2 = bbase[2 * nstride];
    v16h bf3 = bbase[3 * nstride];
    acc[0] = __builtin_amdgcn_wmma_f32_16x16x32_f16(false, af, false, bf0,
                                                    (short)0, acc[0], false, false);
    acc[1] = __builtin_amdgcn_wmma_f32_16x16x32_f16(false, af, false, bf1,
                                                    (short)0, acc[1], false, false);
    acc[2] = __builtin_amdgcn_wmma_f32_16x16x32_f16(false, af, false, bf2,
                                                    (short)0, acc[2], false, false);
    acc[3] = __builtin_amdgcn_wmma_f32_16x16x32_f16(false, af, false, bf3,
                                                    (short)0, acc[3], false, false);
  }

  // C/D layout: vgpr r -> M = r + (lane>=16 ? 8 : 0), N = lane&15
  int mbase = mtile * 16 + (kg << 3);
#pragma unroll
  for (int q = 0; q < 4; ++q) {
    int n0 = (ntile0 + q) * 16 + ml;
    float bv = bias ? bias[n0] : 0.0f;
#pragma unroll
    for (int r = 0; r < 8; ++r) {
      size_t idx = (size_t)(mbase + r) * N + n0;
      float v = acc[q][r] + bv;
      if (accum) v += C[idx];
      if (act == 1) v = (v > 20.0f) ? v : log1pf(__expf(v));
      C[idx] = v;
    }
  }
}

// ---------------------------------------------------------------------------
// LayerNorm over 512 channels (+optional residual, ReLU, and mlp_a dot).
// One block (256 threads) per row; 2 elements per thread.
// ---------------------------------------------------------------------------
__global__ void ln_fuse_kernel(const float* __restrict__ X,
                               const float* __restrict__ R,
                               const float* __restrict__ w,
                               const float* __restrict__ b,
                               float* __restrict__ Y, int relu,
                               const float* __restrict__ aw,
                               const float* __restrict__ ab,
                               float* __restrict__ coff) {
  __shared__ float red[256];
  int row = blockIdx.x, tid = threadIdx.x;
  size_t base = (size_t)row * DIMC;
  float v0 = X[base + tid], v1 = X[base + tid + 256];
  if (R) { v0 += R[base + tid]; v1 += R[base + tid + 256]; }
  red[tid] = v0 + v1;
  __syncthreads();
  for (int o = 128; o > 0; o >>= 1) { if (tid < o) red[tid] += red[tid + o]; __syncthreads(); }
  float mu = red[0] * (1.0f / DIMC);
  __syncthreads();
  float d0 = v0 - mu, d1 = v1 - mu;
  red[tid] = d0 * d0 + d1 * d1;
  __syncthreads();
  for (int o = 128; o > 0; o >>= 1) { if (tid < o) red[tid] += red[tid + o]; __syncthreads(); }
  float rs = rsqrtf(red[0] * (1.0f / DIMC) + 1e-5f);
  __syncthreads();
  float y0 = d0 * rs * w[tid] + b[tid];
  float y1 = d1 * rs * w[tid + 256] + b[tid + 256];
  if (relu) { y0 = fmaxf(y0, 0.0f); y1 = fmaxf(y1, 0.0f); }
  if (Y) { Y[base + tid] = y0; Y[base + tid + 256] = y1; }
  if (coff) {
    red[tid] = y0 * aw[tid] + y1 * aw[tid + 256];
    __syncthreads();
    for (int o = 128; o > 0; o >>= 1) { if (tid < o) red[tid] += red[tid + o]; __syncthreads(); }
    if (tid == 0) coff[row] = red[0] + ab[0];
  }
}

// Depthwise causal conv1d (k=4, left pad 3) + SiLU. x = xz[:, :, :1024].
__global__ void dwconv_silu_kernel(const float* __restrict__ xz,
                                   const float* __restrict__ cw,
                                   const float* __restrict__ cb,
                                   float* __restrict__ xs) {
  int id = blockIdx.x * blockDim.x + threadIdx.x;  // M * D_INNER threads
  int d  = id & (D_INNER - 1);
  int bl = id >> 10;
  int l  = bl & (LSEQ - 1);
  float acc = cb[d];
#pragma unroll
  for (int t = 0; t < D_CONVK; ++t) {
    int ls = l + t - (D_CONVK - 1);
    if (ls >= 0)
      acc += cw[d * D_CONVK + t] * xz[(size_t)(bl + ls - l) * (2 * D_INNER) + d];
  }
  xs[id] = acc / (1.0f + __expf(-acc));
}

// Selective scan: one thread per (batch, channel); 16 states in registers.
__global__ void scan_kernel(const float* __restrict__ dt,
                            const float* __restrict__ xs,
                            const float* __restrict__ dbc,
                            const float* __restrict__ xz,
                            const float* __restrict__ A_log,
                            const float* __restrict__ Dsk,
                            float* __restrict__ g) {
  int id = blockIdx.x * blockDim.x + threadIdx.x;  // BATCH * D_INNER
  int d = id & (D_INNER - 1);
  int b = id >> 10;
  float Ac[D_STATE], h[D_STATE];
#pragma unroll
  for (int n = 0; n < D_STATE; ++n) { Ac[n] = -__expf(A_log[d * D_STATE + n]); h[n] = 0.0f; }
  float Dv = Dsk[d];
  for (int l = 0; l < LSEQ; ++l) {
    size_t bl = (size_t)b * LSEQ + l;
    float dtv = dt[bl * D_INNER + d];
    float xv  = xs[bl * D_INNER + d];
    float zv  = xz[bl * (2 * D_INNER) + D_INNER + d];
    const float* Bm = dbc + bl * (DT_RANK + 2 * D_STATE) + DT_RANK;
    const float* Cm = Bm + D_STATE;
    float y = 0.0f;
#pragma unroll
    for (int n = 0; n < D_STATE; ++n) {
      h[n] = __expf(dtv * Ac[n]) * h[n] + dtv * xv * Bm[n];
      y += h[n] * Cm[n];
    }
    float sz = zv / (1.0f + __expf(-zv));
    g[bl * D_INNER + d] = (y + xv * Dv) * sz;
  }
}

// Per-batch: sigmoid(sum_l coff[b,l]*mlp_b_w[l] + mlp_b_b).
__global__ void final_kernel(const float* __restrict__ coff,
                             const float* __restrict__ bw,
                             const float* __restrict__ bb,
                             float* __restrict__ out) {
  __shared__ float red[128];
  int b = blockIdx.x, tid = threadIdx.x;
  red[tid] = coff[b * LSEQ + tid] * bw[tid];
  __syncthreads();
  for (int o = 64; o > 0; o >>= 1) { if (tid < o) red[tid] += red[tid + o]; __syncthreads(); }
  if (tid == 0) {
    float s = red[0] + bb[0];
    out[b] = 1.0f / (1.0f + __expf(-s));
  }
}

// ---------------------------------------------------------------------------
extern "C" void kernel_launch(void* const* d_in, const int* in_sizes, int n_in,
                              void* d_out, int out_size, void* d_ws, size_t ws_size,
                              hipStream_t stream) {
  (void)in_sizes; (void)n_in; (void)out_size; (void)ws_size;
  const float* spt   = (const float*)d_in[0];
  const float* qry   = (const float*)d_in[1];
  const float* convw = (const float*)d_in[2];
  const float* convb = (const float*)d_in[3];
  const float* lnw   = (const float*)d_in[4];
  const float* lnb   = (const float*)d_in[5];
  const float* inpw  = (const float*)d_in[6];
  const float* c1w   = (const float*)d_in[7];
  const float* c1b   = (const float*)d_in[8];
  const float* xpw   = (const float*)d_in[9];
  const float* dtw   = (const float*)d_in[10];
  const float* dtb   = (const float*)d_in[11];
  const float* alog  = (const float*)d_in[12];
  const float* dskip = (const float*)d_in[13];
  const float* outw  = (const float*)d_in[14];
  const float* maw   = (const float*)d_in[15];
  const float* mab   = (const float*)d_in[16];
  const float* mbw   = (const float*)d_in[17];
  const float* mbb   = (const float*)d_in[18];
  float* out = (float*)d_out;

  const int M = MROWS;
  const int NDBC = DT_RANK + 2 * D_STATE;  // 64

  float* F = (float*)d_ws;
  size_t off = 0;
  auto alloc = [&](size_t n) { float* p = F + off; off += n; return p; };
  float* fts0  = alloc((size_t)M * DIMC);
  float* convC = alloc((size_t)M * DIMC);
  float* u     = alloc((size_t)M * DIMC);
  float* xz    = alloc((size_t)M * 2 * D_INNER);
  float* xs    = alloc((size_t)M * D_INNER);
  float* dbc   = alloc((size_t)M * NDBC);
  float* dtbuf = alloc((size_t)M * D_INNER);
  float* gbuf  = alloc((size_t)M * D_INNER);
  float* mo    = alloc((size_t)M * DIMC);
  float* coff  = alloc((size_t)M);
  _Float16* H = (_Float16*)(F + off);
  size_t hoff = 0;
  auto halloc = [&](size_t n) { _Float16* p = H + hoff; hoff += n; return p; };
  _Float16* pk_conv0 = halloc((size_t)DIMC * DIMC);
  _Float16* pk_conv1 = halloc((size_t)DIMC * DIMC);
  _Float16* pk_conv2 = halloc((size_t)DIMC * DIMC);
  _Float16* pk_in  = halloc((size_t)2 * D_INNER * DIMC);
  _Float16* pk_xp  = halloc((size_t)NDBC * D_INNER);
  _Float16* pk_dt  = halloc((size_t)D_INNER * DT_RANK);
  _Float16* pk_out = halloc((size_t)DIMC * D_INNER);

  concat_kernel<<<(BATCH * 2 * HWC * HWC * DIMC) / 256, 256, 0, stream>>>(spt, qry, fts0);

  auto pack = [&](const float* W, _Float16* Wp, int N, int K, int ldn, int strk) {
    int total = (N >> 4) * (K >> 5) * 32;
    pack_w_kernel<<<(total + 255) / 256, 256, 0, stream>>>(W, Wp, N, K, ldn, strk);
  };
  pack(convw + 0, pk_conv0, DIMC, DIMC, DIMC * 3, 3);
  pack(convw + 1, pk_conv1, DIMC, DIMC, DIMC * 3, 3);
  pack(convw + 2, pk_conv2, DIMC, DIMC, DIMC * 3, 3);
  pack(inpw, pk_in, 2 * D_INNER, DIMC, DIMC, 1);
  pack(xpw,  pk_xp, NDBC, D_INNER, D_INNER, 1);
  pack(dtw,  pk_dt, D_INNER, DT_RANK, DT_RANK, 1);
  pack(outw, pk_out, DIMC, D_INNER, D_INNER, 1);

  auto gemm = [&](const float* A, const _Float16* Wp, float* C, const float* bias,
                  int N, int K, int lda, int shift, int use_seq, int accum, int act) {
    int tiles = (M / 16) * (N / 64);   // one 16x64 strip per wave
    wmma_gemm_kernel<<<tiles / 4, 128, 0, stream>>>(A, Wp, C, bias, M, N, K, lda,
                                                    shift, use_seq, accum, act);
  };

  // conv1d (SAME, k=3) as three accumulating GEMM passes with shifted rows
  gemm(fts0, pk_conv0, convC, nullptr, DIMC, DIMC, DIMC, -1, 1, 0, 0);
  gemm(fts0, pk_conv1, convC, nullptr, DIMC, DIMC, DIMC,  0, 1, 1, 0);
  gemm(fts0, pk_conv2, convC, convb,   DIMC, DIMC, DIMC, +1, 1, 1, 0);

  // LN + ReLU -> u (mamba input & residual)
  ln_fuse_kernel<<<M, 256, 0, stream>>>(convC, nullptr, lnw, lnb, u, 1,
                                        nullptr, nullptr, nullptr);

  // in_proj: (8192 x 2048)
  gemm(u, pk_in, xz, nullptr, 2 * D_INNER, DIMC, DIMC, 0, 0, 0, 0);

  // depthwise causal conv + SiLU
  dwconv_silu_kernel<<<(M * D_INNER) / 256, 256, 0, stream>>>(xz, c1w, c1b, xs);

  // x_proj: (8192 x 64)
  gemm(xs, pk_xp, dbc, nullptr, NDBC, D_INNER, D_INNER, 0, 0, 0, 0);

  // dt_proj + softplus: (8192 x 1024), A = dbc[:, :32] (lda=64)
  gemm(dbc, pk_dt, dtbuf, dtb, D_INNER, DT_RANK, NDBC, 0, 0, 0, 1);

  // selective scan -> g = (y + x*D) * silu(z)
  scan_kernel<<<(BATCH * D_INNER) / 256, 256, 0, stream>>>(dtbuf, xs, dbc, xz,
                                                           alog, dskip, gbuf);

  // out_proj: (8192 x 512)
  gemm(gbuf, pk_out, mo, nullptr, DIMC, D_INNER, D_INNER, 0, 0, 0, 0);

  // residual + LN + mlp_a dot -> coff  (Y reuses fts0, no longer needed)
  ln_fuse_kernel<<<M, 256, 0, stream>>>(mo, u, lnw, lnb, fts0, 0, maw, mab, coff);

  // per-batch mlp_b reduction + sigmoid
  final_kernel<<<BATCH, 128, 0, stream>>>(coff, mbw, mbb, out);
}
